// WindowAttention_16097537426158
// MI455X (gfx1250) — matrix operand order repaired and checked by reading, not compiled, not run on hardware
//
#include <hip/hip_runtime.h>
#include <hip/hip_bf16.h>
#include <stdint.h>

// ---------------------------------------------------------------------------
// Window attention (Swin), MI455X / gfx1250, bf16 WMMA pipeline.
//   DIM=384  WS=7  NH=12  N=49 (padded to 64)  B=4096  HEAD_DIM=32
// ---------------------------------------------------------------------------

#define DIMC   384
#define NTOK   49
#define NPAD   64
#define NHEAD  12
#define HD     32
#define NB     4096
#define SCALEF 0.17677669529663687f   // 32^-0.5

typedef __bf16 bf16_t;
typedef __attribute__((ext_vector_type(16))) __bf16   bf16x16;
typedef __attribute__((ext_vector_type(8)))  float    f32x8;
typedef __attribute__((ext_vector_type(8)))  uint32_t u32x8;

__device__ inline f32x8 splat8(float v) {
    f32x8 r;
#pragma unroll
    for (int i = 0; i < 8; ++i) r[i] = v;
    return r;
}

__device__ inline bf16x16 cast_frag(u32x8 u) {
    union { u32x8 u; bf16x16 b; } cv; cv.u = u; return cv.b;
}

// A fragment, 16-bit 16x32 (code's own k order, not the ISA table's): lane%16 = row M, lane/16 picks
// K subset: k = 2*v + p + 16*half  for element e=2v+p.
// `row` points at this lane's row, already offset to the K-chunk base.
__device__ inline bf16x16 load_a_bf16(const bf16_t* row, int half) {
    u32x8 r;
#pragma unroll
    for (int v = 0; v < 8; ++v) {
        int kb = (v << 1) + (half << 4);   // even
        r[v] = *reinterpret_cast<const uint32_t*>(row + kb);
    }
    return cast_frag(r);
}

// B-matrix fragment, 16-bit 32x16: lane%16 = col N, lanes0-15 K=0..15,
// lanes16-31 K=16..31 (16 contiguous K elements). `p` = 16 contiguous bf16.
__device__ inline bf16x16 load_b_contig(const bf16_t* p) {
    return *reinterpret_cast<const bf16x16*>(p);  // 32B aligned
}

// B fragment where consecutive K elements are `ld` apart (V matrix).
__device__ inline bf16x16 load_b_strided(const bf16_t* p, int ld) {
    bf16x16 f;
#pragma unroll
    for (int e = 0; e < 16; ++e) f[e] = p[e * ld];
    return f;
}

__device__ inline f32x8 wmma_bf16(bf16x16 a, bf16x16 b, f32x8 c) {
    return __builtin_amdgcn_wmma_f32_16x16x32_bf16(false, a, false, b,
                                                   (short)0, c, false, false);
}

// ---------------------------------------------------------------------------
// Kernel W: fp32 -> bf16 weight conversion (runs once per launch; tiny).
// ---------------------------------------------------------------------------
__global__ __launch_bounds__(256) void cvt_kernel(
        const float* __restrict__ src, bf16_t* __restrict__ dst, int n) {
    int i = (blockIdx.x * 256 + threadIdx.x) * 4;
    if (i + 3 < n) {
#pragma unroll
        for (int k = 0; k < 4; ++k) dst[i + k] = (bf16_t)src[i + k];
    } else {
        for (; i < n; ++i) dst[i] = (bf16_t)src[i];
    }
}

// ---------------------------------------------------------------------------
// Kernel 0: padded relative-position bias table  biasp[12][64][64] fp32.
// Key columns >= 49 get -1e30 (softmax mask for the padding).
// ---------------------------------------------------------------------------
__global__ __launch_bounds__(256) void bias_kernel(
        const float* __restrict__ table, const int* __restrict__ rel,
        float* __restrict__ biasp) {
    int idx = blockIdx.x * 256 + threadIdx.x;
    if (idx >= NHEAD * NPAD * NPAD) return;
    int h = idx / (NPAD * NPAD);
    int r = (idx / NPAD) % NPAD;
    int c = idx % NPAD;
    float v;
    if (c >= NTOK)      v = -1e30f;               // mask padded keys
    else if (r >= NTOK) v = 0.0f;                 // padded queries: harmless
    else                v = table[rel[r * NTOK + c] * NHEAD + h];
    biasp[idx] = v;
}

// ---------------------------------------------------------------------------
// Kernel 1: QKV projection, one workgroup (8 waves) per window.
// out layout: qkv[b][head][which(q,k,v)][m 0..63][hd 0..31] bf16, q pre-scaled.
// ---------------------------------------------------------------------------
__global__ __launch_bounds__(256) void qkv_kernel(
        const float* __restrict__ x, const bf16_t* __restrict__ w,
        const float* __restrict__ bvec, bf16_t* __restrict__ qkv) {
    __shared__ bf16_t xs[NPAD][DIMC];                       // 48 KB
    const int b   = blockIdx.x;
    const int tid = threadIdx.x;

    const float* xb = x + (size_t)b * NTOK * DIMC;
    for (int i = tid; i < NPAD * DIMC; i += 256) {
        int row = i / DIMC, col = i % DIMC;
        xs[row][col] = (bf16_t)((row < NTOK) ? xb[row * DIMC + col] : 0.0f);
    }
    __syncthreads();

    const int wave = tid >> 5, lane = tid & 31;
    const int half = lane >> 4, ln = lane & 15;

    for (int t = 0; t < 9; ++t) {                           // 72 col-tiles / 8 waves
        const int ct    = wave * 9 + t;
        const int which = ct / 24;                          // 0=q 1=k 2=v (24 tiles each)
        const int cr    = ct - which * 24;                  // tile within q/k/v block
        const int head  = cr >> 1;
        const int hd0   = (cr & 1) << 4;
        const int j     = ct * 16 + ln;                     // this lane's out col

        const bf16_t* wrow = w + (size_t)j * DIMC + half * 16;

        f32x8 c[4];
#pragma unroll
        for (int mt = 0; mt < 4; ++mt) c[mt] = splat8(0.0f);

        for (int kt = 0; kt < 12; ++kt) {                   // K = 384 / 32
            const int k0 = kt * 32;
            bf16x16 bf = load_b_contig(wrow + k0);
            if (kt + 1 < 12)
                __builtin_prefetch(wrow + k0 + 32, 0, 3);
#pragma unroll
            for (int mt = 0; mt < 4; ++mt) {
                bf16x16 af = load_a_bf16(&xs[mt * 16 + ln][k0], half);
                c[mt] = wmma_bf16(af, bf, c[mt]);
            }
        }

        const float bias  = bvec[j];
        const float scale = (which == 0) ? SCALEF : 1.0f;
        bf16_t* outb = qkv + ((((size_t)b * NHEAD + head) * 3 + which) * NPAD) * HD
                           + hd0 + ln;
#pragma unroll
        for (int mt = 0; mt < 4; ++mt)
#pragma unroll
            for (int r = 0; r < 8; ++r) {
                int m = mt * 16 + r + half * 8;
                outb[(size_t)m * HD] = (bf16_t)((c[mt][r] + bias) * scale);
            }
    }
}

// ---------------------------------------------------------------------------
// Kernel 2: attention per (window, head). 4 waves, wave = one 16-row M-tile.
// S = Qs*K^T (+bias via C-init), register softmax, P@V, bf16 out [b][64][384].
// ---------------------------------------------------------------------------
__global__ __launch_bounds__(128) void attn_kernel(
        const bf16_t* __restrict__ qkv, const float* __restrict__ biasp,
        bf16_t* __restrict__ attn_out) {
    __shared__ bf16_t p_lds[4][16][NPAD];                   // 8 KB
    const int bh = blockIdx.x;
    const int b = bh / NHEAD, h = bh % NHEAD;
    const int tid = threadIdx.x;
    const int wave = tid >> 5, lane = tid & 31;
    const int half = lane >> 4, ln = lane & 15;
    const int mt = wave;

    const bf16_t* qb = qkv + ((size_t)bh * 3) * NPAD * HD;
    const bf16_t* kb = qb + NPAD * HD;
    const bf16_t* vb = qb + 2 * NPAD * HD;

    // A = this wave's Q tile (rows mt*16 .. mt*16+15)
    bf16x16 aq = load_a_bf16(qb + (size_t)(mt * 16 + ln) * HD, half);

    const float* bp = biasp + ((size_t)h * NPAD + mt * 16) * NPAD;
    f32x8 s[4];
#pragma unroll
    for (int nt = 0; nt < 4; ++nt) {
        f32x8 cinit;
#pragma unroll
        for (int r = 0; r < 8; ++r)
            cinit[r] = bp[(r + half * 8) * NPAD + nt * 16 + ln];
        bf16x16 bk = load_b_contig(kb + (size_t)(nt * 16 + ln) * HD + half * 16);
        s[nt] = wmma_bf16(aq, bk, cinit);
    }

    // softmax across the 64 key columns; row m = mt*16 + r + 8*half lives in
    // one 16-lane half -> xor masks 1,2,4,8 stay inside the half.
#pragma unroll
    for (int r = 0; r < 8; ++r) {
        float mx = fmaxf(fmaxf(s[0][r], s[1][r]), fmaxf(s[2][r], s[3][r]));
#pragma unroll
        for (int d = 8; d >= 1; d >>= 1) mx = fmaxf(mx, __shfl_xor(mx, d, 32));
        float sum = 0.0f;
#pragma unroll
        for (int nt = 0; nt < 4; ++nt) {
            float e = __expf(s[nt][r] - mx);
            s[nt][r] = e; sum += e;
        }
#pragma unroll
        for (int d = 8; d >= 1; d >>= 1) sum += __shfl_xor(sum, d, 32);
        float inv = 1.0f / sum;
#pragma unroll
        for (int nt = 0; nt < 4; ++nt) s[nt][r] *= inv;
    }

    // stage P in per-wave LDS to re-shape C-layout -> A-layout
#pragma unroll
    for (int nt = 0; nt < 4; ++nt)
#pragma unroll
        for (int r = 0; r < 8; ++r)
            p_lds[wave][r + half * 8][nt * 16 + ln] = (bf16_t)s[nt][r];

    // O[16x32] = P[16x64] @ V[64x32]
#pragma unroll
    for (int nt2 = 0; nt2 < 2; ++nt2) {
        f32x8 o = splat8(0.0f);
#pragma unroll
        for (int kc = 0; kc < 2; ++kc) {
            bf16x16 ap = load_a_bf16(&p_lds[wave][ln][kc * 32], half);
            bf16x16 bv = load_b_strided(
                vb + (size_t)(kc * 32 + half * 16) * HD + nt2 * 16 + ln, HD);
            o = wmma_bf16(ap, bv, o);
        }
        bf16_t* ob = attn_out + (size_t)b * NPAD * DIMC + h * HD + nt2 * 16 + ln;
#pragma unroll
        for (int r = 0; r < 8; ++r) {
            int m = mt * 16 + r + half * 8;
            ob[(size_t)m * DIMC] = (bf16_t)o[r];
        }
    }
}

// ---------------------------------------------------------------------------
// Kernel 3: output projection, one workgroup (8 waves) per window.
// ---------------------------------------------------------------------------
__global__ __launch_bounds__(256) void proj_kernel(
        const bf16_t* __restrict__ attn_out, const bf16_t* __restrict__ w,
        const float* __restrict__ bvec, float* __restrict__ out) {
    __shared__ bf16_t as[NPAD][DIMC];                       // 48 KB
    const int b = blockIdx.x;
    const int tid = threadIdx.x;

    const uint32_t* src = reinterpret_cast<const uint32_t*>(
        attn_out + (size_t)b * NPAD * DIMC);
    uint32_t* dst = reinterpret_cast<uint32_t*>(&as[0][0]);
    for (int i = tid; i < NPAD * DIMC / 2; i += 256) dst[i] = src[i];
    __syncthreads();

    const int wave = tid >> 5, lane = tid & 31;
    const int half = lane >> 4, ln = lane & 15;

    for (int t = 0; t < 3; ++t) {                           // 24 col-tiles / 8 waves
        const int ct = wave * 3 + t;
        const int j  = ct * 16 + ln;
        const bf16_t* wrow = w + (size_t)j * DIMC + half * 16;

        f32x8 c[4];
#pragma unroll
        for (int mt = 0; mt < 4; ++mt) c[mt] = splat8(bvec[j]);

        for (int kt = 0; kt < 12; ++kt) {
            const int k0 = kt * 32;
            bf16x16 bf = load_b_contig(wrow + k0);
            if (kt + 1 < 12)
                __builtin_prefetch(wrow + k0 + 32, 0, 3);
#pragma unroll
            for (int mt = 0; mt < 4; ++mt) {
                bf16x16 af = load_a_bf16(&as[mt * 16 + ln][k0], half);
                c[mt] = wmma_bf16(af, bf, c[mt]);
            }
        }
#pragma unroll
        for (int mt = 0; mt < 4; ++mt)
#pragma unroll
            for (int r = 0; r < 8; ++r) {
                int m = mt * 16 + r + half * 8;
                if (m < NTOK)
                    out[((size_t)b * NTOK + m) * DIMC + j] = c[mt][r];
            }
    }
}

// ---------------------------------------------------------------------------
extern "C" void kernel_launch(void* const* d_in, const int* in_sizes, int n_in,
                              void* d_out, int out_size, void* d_ws, size_t ws_size,
                              hipStream_t stream) {
    const float* x          = (const float*)d_in[0];
    const float* qkv_w      = (const float*)d_in[1];
    const float* qkv_b      = (const float*)d_in[2];
    const float* proj_w     = (const float*)d_in[3];
    const float* proj_b     = (const float*)d_in[4];
    const float* bias_table = (const float*)d_in[5];
    const int*   rel_index  = (const int*)d_in[6];

    char* ws = (char*)d_ws;
    // [0)          : bias_pad fp32 [12][64][64]            = 196,608 B
    // [196608)     : qkv_w  bf16 [1152][384]               = 884,736 B
    // [1081344)    : proj_w bf16 [384][384]                = 294,912 B
    // [1376256)    : qkv bf16 [B][12][3][64][32]           = 603,979,776 B
    // [605356032)  : attn_out bf16 [B][64][384]            = 201,326,592 B
    float*  biasp    = (float*)ws;
    bf16_t* qkv_w_bf = (bf16_t*)(ws + 196608);
    bf16_t* prj_w_bf = (bf16_t*)(ws + 1081344);
    bf16_t* qkv_ws   = (bf16_t*)(ws + 1376256);
    bf16_t* attn_ws  = (bf16_t*)(ws + 605356032ull);

    const int nq = 3 * DIMC * DIMC, np = DIMC * DIMC;
    cvt_kernel<<<(nq / 4 + 255) / 256, 256, 0, stream>>>(qkv_w, qkv_w_bf, nq);
    cvt_kernel<<<(np / 4 + 255) / 256, 256, 0, stream>>>(proj_w, prj_w_bf, np);
    bias_kernel<<<(NHEAD * NPAD * NPAD + 255) / 256, 256, 0, stream>>>(
        bias_table, rel_index, biasp);
    qkv_kernel<<<NB, 256, 0, stream>>>(x, qkv_w_bf, qkv_b, qkv_ws);
    attn_kernel<<<NB * NHEAD, 128, 0, stream>>>(qkv_ws, biasp, attn_ws);
    proj_kernel<<<NB, 256, 0, stream>>>(attn_ws, prj_w_bf, proj_b, (float*)d_out);
}
